// StyleAttention_17609365914126
// MI455X (gfx1250) — matrix operand order repaired and checked by reading, not compiled, not run on hardware
//
#include <hip/hip_runtime.h>
#include <stdint.h>

typedef __attribute__((ext_vector_type(16))) __bf16 v16bf;
typedef __attribute__((ext_vector_type(8)))  __bf16 v8bf;
typedef __attribute__((ext_vector_type(8)))  float  v8f;

#define DEVINL static __device__ __forceinline__

// ---------------- problem constants ----------------
constexpr int B_  = 8;
constexpr int N_  = 1025;
constexpr int C_  = 1024;
constexpr int H_  = 8;
constexpr int D_  = 128;
constexpr int MROWS = B_ * N_;        // 8200
constexpr int MPAD  = 8256;           // 129 * 64
constexpr int QT_   = 65;             // 16-row q tiles per (b,h)  (65*16 = 1040 >= 1025)
constexpr int KT_   = 66;             // 16-col k tiles per (b,h)  (66*16 = 1056)
constexpr int KC_   = 33;             // 32-key chunks per (b,h)
constexpr float QSCALE = 0.08838834764831845f;  // D^-0.5

DEVINL v8f wmma_bf16(v16bf a, v16bf b, v8f c) {
  return __builtin_amdgcn_wmma_f32_16x16x32_bf16(false, a, false, b, (short)0, c, false, false);
}

DEVINL v16bf join16(v8bf lo, v8bf hi) {
  v16bf r;
#pragma unroll
  for (int j = 0; j < 8; ++j) { r[j] = lo[j]; r[j + 8] = hi[j]; }
  return r;
}

// ---------------------------------------------------------------------------
// Shared block GEMM core: C(64x128) += A(64xK) * B(Kx128), bf16 in, f32 acc.
// 256 threads = 8 waves; wave w covers M rows [(w&1)*32, +32), N cols [(w>>1)*32, +32)
// => 4 accumulator tiles (2x2 of 16x16) per wave, 4 WMMAs per K=32 chunk.
// A row-major [M][K] (lda); B row-major [K][N] (ldb).
// LDS: Abuf 64x32 row-major, Bbuf stored transposed [n(128)][k(32)].
// ---------------------------------------------------------------------------
DEVINL void gemm_tiles(const __bf16* __restrict__ A, int lda,
                       const __bf16* __restrict__ Bm, int ldb,
                       int mBase, int nBase, int kIters,
                       __bf16* Abuf, __bf16* Bbuf, v8f (*acc)[2]) {
  const int tid  = threadIdx.x;
  const int lane = tid & 31;
  const int wave = tid >> 5;
  const int wm   = wave & 1;
  const int wn   = wave >> 1;
  for (int kb = 0; kb < kIters; ++kb) {
    __syncthreads();
    { // A tile 64x32, 16B per thread
      int row = tid >> 2;
      int kc  = (tid & 3) * 8;
      const __bf16* src = A + (long)(mBase + row) * lda + kb * 32 + kc;
      *(uint4*)&Abuf[row * 32 + kc] = *(const uint4*)src;
      if (kb + 1 < kIters) __builtin_prefetch(src + 32, 0, 3);   // global_prefetch_b8
    }
    { // B tile 32x128 -> LDS transposed [n][k]
#pragma unroll
      for (int rep = 0; rep < 2; ++rep) {
        int k  = (tid >> 4) + rep * 16;     // 0..31
        int n0 = (tid & 15) * 8;            // 0..120
        const __bf16* src = Bm + (long)(kb * 32 + k) * ldb + nBase + n0;
        v8bf v = *(const v8bf*)src;
        if (kb + 1 < kIters) __builtin_prefetch(src + 32 * ldb, 0, 3);
#pragma unroll
        for (int j = 0; j < 8; ++j) Bbuf[(n0 + j) * 32 + k] = v[j];
      }
    }
    __syncthreads();
    v16bf a[2], b[2];
#pragma unroll
    for (int amt = 0; amt < 2; ++amt) {
      // A operand: lane L -> M = (L&15); element e: K = e + 16*(L>>4)
      int mrow = wm * 32 + amt * 16 + (lane & 15);
      int koff = (lane >> 4) * 16;
      v8bf alo = *(const v8bf*)&Abuf[mrow * 32 + koff];
      v8bf ahi = *(const v8bf*)&Abuf[mrow * 32 + 8 + koff];
      a[amt] = join16(alo, ahi);
    }
#pragma unroll
    for (int nt = 0; nt < 2; ++nt) {
      // B operand: lane L -> N = (L&15); element e: K = e + 16*(L>>4)
      int ncol = wn * 32 + nt * 16 + (lane & 15);
      int kb2  = (lane >> 4) * 16;
      b[nt] = *(const v16bf*)&Bbuf[ncol * 32 + kb2];
    }
#pragma unroll
    for (int amt = 0; amt < 2; ++amt)
#pragma unroll
      for (int nt = 0; nt < 2; ++nt)
        acc[amt][nt] = wmma_bf16(a[amt], b[nt], acc[amt][nt]);
  }
}

// ---------------------------------------------------------------------------
// fp32 -> bf16 convert
// ---------------------------------------------------------------------------
__global__ void cvt_f32_bf16(const float* __restrict__ s, __bf16* __restrict__ d, int n) {
  int i = blockIdx.x * 256 + threadIdx.x;
  if (i < n) d[i] = (__bf16)s[i];
}

// tokens^T[c][t] = sum_l a[t,l] * b[l,c]   (a: 1024x16, b: 16x1024)
__global__ void tokens_kernel(const float* __restrict__ a, const float* __restrict__ b,
                              __bf16* __restrict__ out) {
  int i = blockIdx.x * 256 + threadIdx.x;      // i = c*1024 + t
  if (i >= 1024 * 1024) return;
  int c = i >> 10, t = i & 1023;
  float acc = 0.f;
#pragma unroll
  for (int l = 0; l < 16; ++l) acc += a[t * 16 + l] * b[l * 1024 + c];
  out[i] = (__bf16)acc;
}

// xt[b][c][t] = x[b][1+t][c]   (bf16), tiled LDS transpose
__global__ __launch_bounds__(256) void transpose_cvt(const float* __restrict__ x,
                                                     __bf16* __restrict__ xt) {
  __shared__ float tile[64][65];
  int b = blockIdx.z, cb = blockIdx.x * 64, tb = blockIdx.y * 64;
  for (int idx = threadIdx.x; idx < 4096; idx += 256) {
    int i = idx >> 6, j = idx & 63;          // i: t, j: c
    tile[i][j] = x[((long)b * N_ + 1 + tb + i) * C_ + cb + j];
  }
  __syncthreads();
  for (int idx = threadIdx.x; idx < 4096; idx += 256) {
    int i = idx >> 6, j = idx & 63;          // i: c, j: t
    xt[((long)b * C_ + cb + i) * 1024 + tb + j] = (__bf16)tile[j][i];
  }
}

// ---------------------------------------------------------------------------
// Fused dual GEMM: cols [0,1024)=q, [1024,2048)=kk=k+lora_k, [2048,3072)=vv=v+lora_v.
// Epilogue scatters into WMMA operand-ready bf16 layouts + fp32 vv output.
// ---------------------------------------------------------------------------
__global__ __launch_bounds__(256) void gemm_qkv_kernel(
    const __bf16* __restrict__ xb, const __bf16* __restrict__ skb,
    const __bf16* __restrict__ svb, const __bf16* __restrict__ wqkv,
    const __bf16* __restrict__ tkT, const __bf16* __restrict__ tvT,
    __bf16* __restrict__ qA, __bf16* __restrict__ kkB, __bf16* __restrict__ vvB,
    float* __restrict__ vvout) {
  __shared__ __attribute__((aligned(32))) __bf16 Abuf[64 * 32];
  __shared__ __attribute__((aligned(32))) __bf16 Bbuf[128 * 32];
  int mBase = blockIdx.x * 64;
  int nBase = blockIdx.y * 128;
  int sec   = nBase >> 10;                    // 0:q 1:kk 2:vv (uniform per block)
  v8f acc[2][2] = {{ v8f{0,0,0,0,0,0,0,0}, v8f{0,0,0,0,0,0,0,0} },
                   { v8f{0,0,0,0,0,0,0,0}, v8f{0,0,0,0,0,0,0,0} }};

  gemm_tiles(xb, C_, wqkv, 3 * C_, mBase, nBase, C_ / 32, Abuf, Bbuf, acc);
  if (sec == 1)      gemm_tiles(skb, C_, tkT, C_, mBase, nBase - 1024, C_ / 32, Abuf, Bbuf, acc);
  else if (sec == 2) gemm_tiles(svb, C_, tvT, C_, mBase, nBase - 2048, C_ / 32, Abuf, Bbuf, acc);

  const int lane = threadIdx.x & 31;
  const int wave = threadIdx.x >> 5;
  const int wm = wave & 1, wn = wave >> 1;
#pragma unroll
  for (int amt = 0; amt < 2; ++amt)
#pragma unroll
    for (int nt = 0; nt < 2; ++nt)
#pragma unroll
      for (int r = 0; r < 8; ++r) {
        int rg = mBase + wm * 32 + amt * 16 + r + ((lane >> 4) << 3);
        if (rg >= MROWS) continue;
        float val = acc[amt][nt][r];
        int cg = nBase + wn * 32 + nt * 16 + (lane & 15);
        int b = rg / N_, n = rg % N_;
        int cc = cg & 1023, h = cc >> 7, d = cc & 127;
        int bh = b * H_ + h;
        if (sec == 0) {                          // q -> A layout, fold scale
          val *= QSCALE;
          int qt = n >> 4, nl = n & 15, ch = d >> 5, Kl = d & 31;
          int Lh = Kl >> 4;
          int e  = Kl & 15;
          int L  = Lh * 16 + nl;
          qA[(((long)(bh * QT_ + qt)) * 4 + ch) * 512 + L * 16 + e] = (__bf16)val;
        } else if (sec == 1) {                   // kk -> B layout (K = d)
          int kt = n >> 4, nl = n & 15, ch = d >> 5, Kl = d & 31;
          int L = ((Kl >> 4) << 4) + nl, e = Kl & 15;
          kkB[(((long)(bh * KT_ + kt)) * 4 + ch) * 512 + L * 16 + e] = (__bf16)val;
        } else {                                 // vv: fp32 output + B layout (K = key)
          vvout[(long)rg * C_ + cc] = val;
          int kc = n >> 5, Kl = n & 31, dt = d >> 4, nl2 = d & 15;
          int L = ((Kl >> 4) << 4) + nl2, e = Kl & 15;
          vvB[(((long)(bh * KC_ + kc)) * 8 + dt) * 512 + L * 16 + e] = (__bf16)val;
        }
      }
}

// gating GEMM1: h = relu(xt[b] @ W1 + b1), per batch, M=1024 N=256 K=1024
__global__ __launch_bounds__(256) void gemm_gate1_kernel(
    const __bf16* __restrict__ xt, const __bf16* __restrict__ w1b,
    const float* __restrict__ b1, __bf16* __restrict__ hb) {
  __shared__ __attribute__((aligned(32))) __bf16 Abuf[64 * 32];
  __shared__ __attribute__((aligned(32))) __bf16 Bbuf[128 * 32];
  int b = blockIdx.z;
  int mBase = blockIdx.x * 64, nBase = blockIdx.y * 128;
  v8f acc[2][2] = {{ v8f{0,0,0,0,0,0,0,0}, v8f{0,0,0,0,0,0,0,0} },
                   { v8f{0,0,0,0,0,0,0,0}, v8f{0,0,0,0,0,0,0,0} }};
  gemm_tiles(xt + (long)b * C_ * 1024, 1024, w1b, 256, mBase, nBase, 1024 / 32, Abuf, Bbuf, acc);
  const int lane = threadIdx.x & 31;
  const int wave = threadIdx.x >> 5;
  const int wm = wave & 1, wn = wave >> 1;
#pragma unroll
  for (int amt = 0; amt < 2; ++amt)
#pragma unroll
    for (int nt = 0; nt < 2; ++nt)
#pragma unroll
      for (int r = 0; r < 8; ++r) {
        int rg = mBase + wm * 32 + amt * 16 + r + ((lane >> 4) << 3);
        int cg = nBase + wn * 32 + nt * 16 + (lane & 15);
        float v = fmaxf(acc[amt][nt][r] + b1[cg], 0.f);
        hb[((long)b * 1024 + rg) * 256 + cg] = (__bf16)v;
      }
}

// gating tail: wcls[b][c] = x[b,0,c] * sigmoid(h[b,c,:] . W2 + b2)
__global__ void gate2_kernel(const __bf16* __restrict__ hb, const float* __restrict__ W2,
                             const float* __restrict__ b2, const float* __restrict__ x,
                             float* __restrict__ wcls) {
  int i = blockIdx.x * 256 + threadIdx.x;     // i = b*1024 + c
  if (i >= B_ * C_) return;
  int b = i >> 10, c = i & 1023;
  const __bf16* hrow = hb + (long)i * 256;
  float acc = b2[0];
  for (int j = 0; j < 256; ++j) acc += (float)hrow[j] * W2[j];
  float sg = 1.f / (1.f + __expf(-acc));
  wcls[i] = x[((long)b * N_) * C_ + c] * sg;
}

// ---------------------------------------------------------------------------
// Flash attention: grid (B*H, ceil(QT_/8)), 256 threads = 8 waves, one 16-query
// tile per wave. K/V chunks staged to LDS with GLOBAL_LOAD_ASYNC_TO_LDS_B128
// (ASYNCcnt path); online softmax; operands pre-swizzled (qA/kkB/vvB).
// ---------------------------------------------------------------------------
__global__ __launch_bounds__(256) void attn_kernel(const __bf16* __restrict__ qA,
                                                   const __bf16* __restrict__ kkB,
                                                   const __bf16* __restrict__ vvB,
                                                   __bf16* __restrict__ attnob) {
  __shared__ __attribute__((aligned(32))) __bf16 kkS[8 * 512];   // 2 ktiles x 4 d-chunks
  __shared__ __attribute__((aligned(32))) __bf16 vvS[8 * 512];   // 8 d-tiles
  __shared__ __attribute__((aligned(32))) __bf16 probs[8][512];  // per-wave 16x32
  const int bh   = blockIdx.x;
  const int wave = threadIdx.x >> 5;
  const int lane = threadIdx.x & 31;
  const int qt   = blockIdx.y * 8 + wave;
  const int qtc  = qt < QT_ - 1 ? qt : QT_ - 1;

  // resident q A-operands (16 x 128)
  v16bf qa[4];
  {
    const __bf16* qb = qA + (((long)(bh * QT_ + qtc)) * 4) * 512;
#pragma unroll
    for (int ch = 0; ch < 4; ++ch)
      qa[ch] = *(const v16bf*)(qb + ch * 512 + lane * 16);
  }
  float m[8], l[8];
  v8f O[8];
  v8f z8 = {0.f, 0.f, 0.f, 0.f, 0.f, 0.f, 0.f, 0.f};
#pragma unroll
  for (int r = 0; r < 8; ++r) { m[r] = -1e30f; l[r] = 0.f; }
#pragma unroll
  for (int dt = 0; dt < 8; ++dt) O[dt] = z8;

  const __bf16* kkbh = kkB + (long)bh * KT_ * 4 * 512;
  const __bf16* vvbh = vvB + (long)bh * KC_ * 8 * 512;
  const unsigned ldsKK = (unsigned)(uintptr_t)&kkS[0];
  const unsigned ldsVV = (unsigned)(uintptr_t)&vvS[0];

  for (int kc = 0; kc < KC_; ++kc) {
    __syncthreads();
    { // stage 8KB K + 8KB V via async global->LDS (TENSOR-free ASYNCcnt path)
      const char* gk = (const char*)(kkbh + (long)(2 * kc) * 4 * 512);
      const char* gv = (const char*)(vvbh + (long)kc * 8 * 512);
      for (int i = threadIdx.x; i < 512; i += 256) {
        unsigned long long ak_ = (unsigned long long)(gk + i * 16);
        unsigned long long av_ = (unsigned long long)(gv + i * 16);
        asm volatile("global_load_async_to_lds_b128 %0, %1, off"
                     :: "v"(ldsKK + i * 16), "v"(ak_) : "memory");
        asm volatile("global_load_async_to_lds_b128 %0, %1, off"
                     :: "v"(ldsVV + i * 16), "v"(av_) : "memory");
      }
      asm volatile("s_wait_asynccnt 0x0" ::: "memory");
    }
    __syncthreads();
    // scores for 32 keys (2 tiles of 16)
    v8f sc[2] = { z8, z8 };
#pragma unroll
    for (int t = 0; t < 2; ++t)
#pragma unroll
      for (int ch = 0; ch < 4; ++ch) {
        v16bf bv = *(const v16bf*)(kkS + (t * 4 + ch) * 512 + lane * 16);
        sc[t] = wmma_bf16(qa[ch], bv, sc[t]);
      }
    // online softmax, probs -> LDS
    int k0 = kc * 32 + (lane & 15);
#pragma unroll
    for (int r = 0; r < 8; ++r) {
      float s0 = (k0      < N_) ? sc[0][r] : -1e30f;
      float s1 = (k0 + 16 < N_) ? sc[1][r] : -1e30f;
      float mx = fmaxf(s0, s1);
#pragma unroll
      for (int off = 1; off < 16; off <<= 1) mx = fmaxf(mx, __shfl_xor(mx, off, 32));
      float mn = fmaxf(m[r], mx);
      float corr = __expf(m[r] - mn);
      m[r] = mn;
      float p0 = __expf(s0 - mn);
      float p1 = __expf(s1 - mn);
      float rs = p0 + p1;
#pragma unroll
      for (int off = 1; off < 16; off <<= 1) rs += __shfl_xor(rs, off, 32);
      l[r] = l[r] * corr + rs;
#pragma unroll
      for (int dt = 0; dt < 8; ++dt) O[dt][r] *= corr;
      int row = r + ((lane >> 4) << 3);
      probs[wave][row * 32 + (lane & 15)]      = (__bf16)p0;
      probs[wave][row * 32 + 16 + (lane & 15)] = (__bf16)p1;
    }
    asm volatile("s_wait_dscnt 0x0" ::: "memory");
    // probs as A operand, then P @ V
    int mrow = lane & 15;
    int koff = (lane >> 4) * 16;
    v8bf plo = *(const v8bf*)&probs[wave][mrow * 32 + koff];
    v8bf phi = *(const v8bf*)&probs[wave][mrow * 32 + 8 + koff];
    v16bf pa = join16(plo, phi);
#pragma unroll
    for (int dt = 0; dt < 8; ++dt) {
      v16bf bv = *(const v16bf*)(vvS + dt * 512 + lane * 16);
      O[dt] = wmma_bf16(pa, bv, O[dt]);
    }
  }
  // epilogue: O / l -> attnob (bf16 row-major [B*N][C])
  int b = bh >> 3, h = bh & 7;
#pragma unroll
  for (int r = 0; r < 8; ++r) {
    int qg = qt * 16 + r + ((lane >> 4) << 3);
    if (qg >= N_) continue;
    float inv = 1.f / l[r];
    long rowoff = ((long)(b * N_ + qg)) * C_ + h * D_;
#pragma unroll
    for (int dt = 0; dt < 8; ++dt)
      attnob[rowoff + dt * 16 + (lane & 15)] = (__bf16)(O[dt][r] * inv);
  }
}

// projection GEMM: out = attno @ W_proj + b_proj (+ wcls on row n==0), fp32 out
__global__ __launch_bounds__(256) void gemm_proj_kernel(
    const __bf16* __restrict__ attnob, const __bf16* __restrict__ wproj,
    const float* __restrict__ b_proj, const float* __restrict__ wcls,
    float* __restrict__ out) {
  __shared__ __attribute__((aligned(32))) __bf16 Abuf[64 * 32];
  __shared__ __attribute__((aligned(32))) __bf16 Bbuf[128 * 32];
  int mBase = blockIdx.x * 64, nBase = blockIdx.y * 128;
  v8f acc[2][2] = {{ v8f{0,0,0,0,0,0,0,0}, v8f{0,0,0,0,0,0,0,0} },
                   { v8f{0,0,0,0,0,0,0,0}, v8f{0,0,0,0,0,0,0,0} }};
  gemm_tiles(attnob, C_, wproj, C_, mBase, nBase, C_ / 32, Abuf, Bbuf, acc);
  const int lane = threadIdx.x & 31;
  const int wave = threadIdx.x >> 5;
  const int wm = wave & 1, wn = wave >> 1;
#pragma unroll
  for (int amt = 0; amt < 2; ++amt)
#pragma unroll
    for (int nt = 0; nt < 2; ++nt)
#pragma unroll
      for (int r = 0; r < 8; ++r) {
        int rg = mBase + wm * 32 + amt * 16 + r + ((lane >> 4) << 3);
        if (rg >= MROWS) continue;
        int cg = nBase + wn * 32 + nt * 16 + (lane & 15);
        int b = rg / N_, n = rg % N_;
        float v = acc[amt][nt][r] + b_proj[cg];
        if (n == 0) v += wcls[b * C_ + cg];
        out[(long)rg * C_ + cg] = v;
      }
}

// ---------------------------------------------------------------------------
extern "C" void kernel_launch(void* const* d_in, const int* in_sizes, int n_in,
                              void* d_out, int out_size, void* d_ws, size_t ws_size,
                              hipStream_t stream) {
  (void)in_sizes; (void)n_in; (void)out_size; (void)ws_size;
  const float* x       = (const float*)d_in[0];
  const float* style_k = (const float*)d_in[1];
  const float* style_v = (const float*)d_in[2];
  const float* W_qkv   = (const float*)d_in[3];
  const float* W_proj  = (const float*)d_in[4];
  const float* b_proj  = (const float*)d_in[5];
  const float* ak      = (const float*)d_in[6];
  const float* bk      = (const float*)d_in[7];
  const float* av      = (const float*)d_in[8];
  const float* bv      = (const float*)d_in[9];
  const float* W1      = (const float*)d_in[10];
  const float* b1      = (const float*)d_in[11];
  const float* W2      = (const float*)d_in[12];
  const float* b2      = (const float*)d_in[13];

  float* out   = (float*)d_out;
  float* vvout = out + (size_t)MROWS * C_;   // second tuple output, same size

  char* ws = (char*)d_ws;
  size_t o = 0;
  auto take = [&](size_t bytes) -> char* {
    char* p = ws + o;
    o = (o + bytes + 255) & ~(size_t)255;
    return p;
  };
  __bf16* xb     = (__bf16*)take((size_t)MPAD * C_ * 2);
  __bf16* skb    = (__bf16*)take((size_t)MPAD * C_ * 2);
  __bf16* svb    = (__bf16*)take((size_t)MPAD * C_ * 2);
  __bf16* wqkvb  = (__bf16*)take((size_t)C_ * 3 * C_ * 2);
  __bf16* wprojb = (__bf16*)take((size_t)C_ * C_ * 2);
  __bf16* w1b    = (__bf16*)take((size_t)C_ * 256 * 2);
  __bf16* tkT    = (__bf16*)take((size_t)C_ * C_ * 2);
  __bf16* tvT    = (__bf16*)take((size_t)C_ * C_ * 2);
  __bf16* xt     = (__bf16*)take((size_t)B_ * C_ * 1024 * 2);
  __bf16* hb     = (__bf16*)take((size_t)B_ * 1024 * 256 * 2);
  float*  wcls   = (float*) take((size_t)B_ * C_ * 4);
  __bf16* qA     = (__bf16*)take((size_t)B_ * H_ * QT_ * 4 * 512 * 2);
  __bf16* kkB    = (__bf16*)take((size_t)B_ * H_ * KT_ * 4 * 512 * 2);
  __bf16* vvB    = (__bf16*)take((size_t)B_ * H_ * KC_ * 8 * 512 * 2);
  __bf16* attnob = (__bf16*)take((size_t)MPAD * C_ * 2);

  auto cvt = [&](const float* s, __bf16* d, int n) {
    cvt_f32_bf16<<<(n + 255) / 256, 256, 0, stream>>>(s, d, n);
  };
  cvt(x,       xb,     MROWS * C_);
  cvt(style_k, skb,    MROWS * C_);
  cvt(style_v, svb,    MROWS * C_);
  cvt(W_qkv,   wqkvb,  C_ * 3 * C_);
  cvt(W_proj,  wprojb, C_ * C_);
  cvt(W1,      w1b,    C_ * 256);

  tokens_kernel<<<(1024 * 1024) / 256, 256, 0, stream>>>(ak, bk, tkT);
  tokens_kernel<<<(1024 * 1024) / 256, 256, 0, stream>>>(av, bv, tvT);

  transpose_cvt<<<dim3(16, 16, 8), 256, 0, stream>>>(x, xt);

  gemm_qkv_kernel<<<dim3(MPAD / 64, 3 * C_ / 128), 256, 0, stream>>>(
      xb, skb, svb, wqkvb, tkT, tvT, qA, kkB, vvB, vvout);

  gemm_gate1_kernel<<<dim3(16, 2, 8), 256, 0, stream>>>(xt, w1b, b1, hb);
  gate2_kernel<<<(B_ * C_) / 256, 256, 0, stream>>>(hb, W2, b2, x, wcls);

  attn_kernel<<<dim3(B_ * H_, (QT_ + 7) / 8), 256, 0, stream>>>(qA, kkB, vvB, attnob);

  gemm_proj_kernel<<<dim3(MPAD / 64, C_ / 128), 256, 0, stream>>>(attnob, wprojb, b_proj, wcls, out);
}
